// Detection_41077067219173
// MI455X (gfx1250) — matrix operand order repaired and checked by reading, not compiled, not run on hardware
//
#include <hip/hip_runtime.h>
#include <hip/hip_bf16.h>
#include <math.h>

// ---------------------------------------------------------------------------
// Problem constants
// ---------------------------------------------------------------------------
#define BB      32
#define TT      256
#define A1      7
#define HD      32
#define DIMX    32
#define HEADS   4
#define IN_FEAT 1014            // 169*6
#define KPAD    1024            // fc1 K padded to multiple of 32
#define NSELF   (BB*TT)         // 8192 self rows
#define NOTHER  (BB*TT*A1)      // 57344 other rows
#define NTILES_SELF  (NSELF/16)     // 512
#define NTILES_OTHER (NOTHER/16)    // 3584
#define NTILES       (NTILES_SELF + NTILES_OTHER)  // 4096
#define FEAT_STRIDE  1032       // halfs per feature row in LDS (bank-conflict pad)

typedef _Float16 v16h __attribute__((ext_vector_type(16)));
typedef _Float16 v8h  __attribute__((ext_vector_type(8)));
typedef float    v8f  __attribute__((ext_vector_type(8)));

static __device__ __forceinline__ v8f wmma_f16(v16h a, v16h b, v8f c) {
    // D(f32,16x16) = A(f16,16x32) x B(f16,32x16) + C
    return __builtin_amdgcn_wmma_f32_16x16x32_f16(
        /*neg_a=*/false, a, /*neg_b=*/false, b,
        /*c_mod=*/(short)0, c, /*reuse_a=*/false, /*reuse_b=*/false);
}

static __device__ __forceinline__ float leaky(float x) {
    return x > 0.f ? x : 0.01f * x;
}

// ---------------------------------------------------------------------------
// Kernel 0: convert weights to f16 (fc1 padded K=1014 -> 1024 with zeros)
// ---------------------------------------------------------------------------
__global__ void prep_weights(const float* __restrict__ fc1_w,
                             const float* __restrict__ fc2_w,
                             const float* __restrict__ key_w,
                             const float* __restrict__ sel_w,
                             _Float16* __restrict__ W1h,   // [32][1024]
                             _Float16* __restrict__ W2h,   // [32][32]
                             _Float16* __restrict__ Keyh,  // [32][32]
                             _Float16* __restrict__ Selh)  // [32][32]
{
    int i = blockIdx.x * blockDim.x + threadIdx.x;
    if (i < 32 * KPAD) {
        int n = i >> 10, k = i & (KPAD - 1);
        W1h[i] = (k < IN_FEAT) ? (_Float16)fc1_w[n * IN_FEAT + k] : (_Float16)0.f;
    }
    if (i < 32 * 32) {
        W2h[i]  = (_Float16)fc2_w[i];
        Keyh[i] = (_Float16)key_w[i];
        Selh[i] = (_Float16)sel_w[i];
    }
}

// ---------------------------------------------------------------------------
// Kernel 1: fused conv -> fc1(WMMA) -> fc2(WMMA) -> key/sel proj (WMMA)
// One wave (32 threads) per tile of 16 samples. Tiles 0..511 = self rows,
// tiles 512..4095 = other rows.
// ---------------------------------------------------------------------------
__global__ __launch_bounds__(32)
void fused_embed(const float* __restrict__ self_state,
                 const float* __restrict__ other_state,
                 const float* __restrict__ conv_w,  // [6][6][3][3] OIHW
                 const float* __restrict__ conv_b,  // [6]
                 const _Float16* __restrict__ W1h,  // [32][1024]
                 const float* __restrict__ fc1_b,
                 const _Float16* __restrict__ W2h,  // [32][32]
                 const float* __restrict__ fc2_b,
                 const _Float16* __restrict__ Keyh, // [32][32]
                 const float* __restrict__ key_b,
                 const _Float16* __restrict__ Selh, // [32][32]
                 const float* __restrict__ sel_b,
                 float* __restrict__ Kbuf,          // [8192][32]
                 float* __restrict__ Sbuf)          // [57344][32]
{
    __shared__ __align__(16) float    w_lds[324];           // [ky][kx][ci][c]
    __shared__ __align__(16) float    in_lds[15 * 15 * 6];  // zero-padded sample
    __shared__ __align__(16) _Float16 feat[16 * FEAT_STRIDE];
    __shared__ __align__(16) _Float16 emb[16 * 40];         // 16x32 act tile, stride 40

    const int  lane   = threadIdx.x;          // 0..31
    const int  tile   = blockIdx.x;           // 0..4095
    const bool isSelf = tile < NTILES_SELF;
    const int  row0   = isSelf ? tile * 16 : (tile - NTILES_SELF) * 16;
    const float* src  = isSelf ? self_state : other_state;

    // conv weights, transposed so output channel c is contiguous (broadcast reads)
    for (int i = lane; i < 324; i += 32) {
        int c = i % 6, t = i / 6, ci = t % 6, kk = t / 6;
        int ky = kk / 3, kx = kk % 3;
        w_lds[i] = conv_w[((c * 6 + ci) * 3 + ky) * 3 + kx];
    }
    float cb[6];
#pragma unroll
    for (int c = 0; c < 6; ++c) cb[c] = conv_b[c];
    __syncthreads();

    // ---------------- conv: 16 samples -> f16 features in LDS ---------------
    for (int r = 0; r < 16; ++r) {
        const float* sp = src + (size_t)(row0 + r) * IN_FEAT;
        if (r < 15) __builtin_prefetch(sp + IN_FEAT, 0, 0);  // global_prefetch_b8

        for (int i = lane; i < 15 * 15 * 6; i += 32) in_lds[i] = 0.f;
        __syncthreads();
        for (int i = lane; i < IN_FEAT; i += 32) {
            float v = sp[i];                    // [13][13][6] row-major
            int c = i % 6, p = i / 6, x = p % 13, y = p / 13;
            in_lds[((y + 1) * 15 + (x + 1)) * 6 + c] = v;
        }
        __syncthreads();

        for (int pp = lane; pp < 169; pp += 32) {
            int y = pp / 13, x = pp % 13;
            float acc[6];
#pragma unroll
            for (int c = 0; c < 6; ++c) acc[c] = cb[c];
            const float* ib = &in_lds[(y * 15 + x) * 6];
            int wi = 0;
#pragma unroll
            for (int ky = 0; ky < 3; ++ky)
#pragma unroll
                for (int kx = 0; kx < 3; ++kx) {
                    const float* pv = ib + (ky * 15 + kx) * 6;
#pragma unroll
                    for (int ci = 0; ci < 6; ++ci) {
                        float v = pv[ci];
                        const float* wr = &w_lds[wi]; wi += 6;
#pragma unroll
                        for (int c = 0; c < 6; ++c) acc[c] += v * wr[c];
                    }
                }
#pragma unroll
            for (int c = 0; c < 6; ++c)
                feat[r * FEAT_STRIDE + c * 169 + pp] = (_Float16)acc[c];
        }
        if (lane < FEAT_STRIDE - IN_FEAT)       // zero K padding 1014..1031
            feat[r * FEAT_STRIDE + IN_FEAT + lane] = (_Float16)0.f;
        __syncthreads();
    }

    // ---------------- fc1: [16x1024] x [1024x32] via WMMA -------------------
    const int rowA = lane & 15;      // A-matrix row (M) for this lane
    const int kg   = lane >> 4;      // K-half select: contiguous k = 16*kg + s in A and B
    const int nlo  = lane & 15;      // D column for c0
    const int nhi  = 16 + nlo;       // D column for c1
    const int mbase = (lane >> 4) * 8;

    v8f c0 = {}; v8f c1 = {};
    const _Float16* frow = &feat[rowA * FEAT_STRIDE];
    for (int k0 = 0; k0 < KPAD; k0 += 32) {
        v8h lo = *(const v8h*)(frow + k0 + kg * 16);
        v8h hi = *(const v8h*)(frow + k0 + kg * 16 + 8);
        v16h a = __builtin_shufflevector(lo, hi, 0,1,2,3,4,5,6,7,8,9,10,11,12,13,14,15);
        v16h b0 = *(const v16h*)(W1h + nlo * KPAD + k0 + kg * 16);
        v16h b1 = *(const v16h*)(W1h + nhi * KPAD + k0 + kg * 16);
        c0 = wmma_f16(a, b0, c0);
        c1 = wmma_f16(a, b1, c1);
    }
    {   // bias + leaky, D-layout -> A-layout via LDS
        float bl = fc1_b[nlo], bh = fc1_b[nhi];
#pragma unroll
        for (int rr = 0; rr < 8; ++rr) {
            emb[(mbase + rr) * 40 + nlo] = (_Float16)leaky(c0[rr] + bl);
            emb[(mbase + rr) * 40 + nhi] = (_Float16)leaky(c1[rr] + bh);
        }
        __syncthreads();
    }

    // ---------------- fc2: [16x32] x [32x32], single K step -----------------
    {
        v8h lo = *(const v8h*)(&emb[rowA * 40 + kg * 16]);
        v8h hi = *(const v8h*)(&emb[rowA * 40 + kg * 16 + 8]);
        v16h a = __builtin_shufflevector(lo, hi, 0,1,2,3,4,5,6,7,8,9,10,11,12,13,14,15);
        __syncthreads();                       // all reads done before overwrite
        v16h b0 = *(const v16h*)(W2h + nlo * 32 + kg * 16);
        v16h b1 = *(const v16h*)(W2h + nhi * 32 + kg * 16);
        v8f d0 = {}; v8f d1 = {};
        d0 = wmma_f16(a, b0, d0);
        d1 = wmma_f16(a, b1, d1);
        float bl = fc2_b[nlo], bh = fc2_b[nhi];
#pragma unroll
        for (int rr = 0; rr < 8; ++rr) {
            emb[(mbase + rr) * 40 + nlo] = (_Float16)leaky(d0[rr] + bl);
            emb[(mbase + rr) * 40 + nhi] = (_Float16)leaky(d1[rr] + bh);
        }
        __syncthreads();
    }

    // ---------------- key / sel projection, single K step -------------------
    {
        const _Float16* Wp = isSelf ? Keyh : Selh;
        const float*    bp = isSelf ? key_b : sel_b;
        v8h lo = *(const v8h*)(&emb[rowA * 40 + kg * 16]);
        v8h hi = *(const v8h*)(&emb[rowA * 40 + kg * 16 + 8]);
        v16h a = __builtin_shufflevector(lo, hi, 0,1,2,3,4,5,6,7,8,9,10,11,12,13,14,15);
        v16h b0 = *(const v16h*)(Wp + nlo * 32 + kg * 16);
        v16h b1 = *(const v16h*)(Wp + nhi * 32 + kg * 16);
        v8f d0 = {}; v8f d1 = {};
        d0 = wmma_f16(a, b0, d0);
        d1 = wmma_f16(a, b1, d1);
        float bl = bp[nlo], bh = bp[nhi];
        float* outb = (isSelf ? Kbuf : Sbuf) + (size_t)row0 * 32;
#pragma unroll
        for (int rr = 0; rr < 8; ++rr) {
            outb[(size_t)(mbase + rr) * 32 + nlo] = d0[rr] + bl;
            outb[(size_t)(mbase + rr) * 32 + nhi] = d1[rr] + bh;
        }
    }
}

// ---------------------------------------------------------------------------
// Kernel 2: scores + 4-head Gumbel softmax + Threat + internal reward
// One thread per (b,t) pair (8192 total).
// ---------------------------------------------------------------------------
__global__ void score_kernel(const float* __restrict__ Kbuf,   // [8192][32]
                             const float* __restrict__ Sbuf,   // [57344][32]
                             const float* __restrict__ diff,   // [8192][7]
                             const float* __restrict__ gumbel, // [4][8192][7]
                             float* __restrict__ out)          // [8192] ++ [8192*7]
{
    int bt = blockIdx.x * blockDim.x + threadIdx.x;
    if (bt >= NSELF) return;

    const float SCALE = 16.0f * sqrtf((float)(A1 * BB * TT * DIMX));
    const float* kv = Kbuf + (size_t)bt * 32;

    float score[A1];
#pragma unroll
    for (int a = 0; a < A1; ++a) {
        const float* sv = Sbuf + ((size_t)bt * A1 + a) * 32;
        float d = 0.f;
#pragma unroll
        for (int j = 0; j < 32; ++j) d += kv[j] * sv[j];
        score[a] = d / SCALE;
    }

    float threat[A1];
#pragma unroll
    for (int a = 0; a < A1; ++a) threat[a] = 0.f;

#pragma unroll
    for (int h = 0; h < HEADS; ++h) {
        const float* g = gumbel + (size_t)h * NSELF * A1 + (size_t)bt * A1;
        float l[A1], m = -3.4e38f;
#pragma unroll
        for (int a = 0; a < A1; ++a) { l[a] = score[a] + g[a]; m = fmaxf(m, l[a]); }
        float s = 0.f;
#pragma unroll
        for (int a = 0; a < A1; ++a) { l[a] = expf(l[a] - m); s += l[a]; }
        float inv = 1.0f / (s * (float)HEADS);
#pragma unroll
        for (int a = 0; a < A1; ++a) threat[a] += l[a] * inv;
    }

    float reward = 0.f;
    const float* dv = diff + (size_t)bt * A1;
#pragma unroll
    for (int a = 0; a < A1; ++a) reward += threat[a] * dv[a];

    out[bt] = reward;
    float* to = out + NSELF + (size_t)bt * A1;
#pragma unroll
    for (int a = 0; a < A1; ++a) to[a] = threat[a];
}

// ---------------------------------------------------------------------------
// Launcher
// ---------------------------------------------------------------------------
extern "C" void kernel_launch(void* const* d_in, const int* in_sizes, int n_in,
                              void* d_out, int out_size, void* d_ws, size_t ws_size,
                              hipStream_t stream) {
    const float* self_state  = (const float*)d_in[0];
    const float* other_state = (const float*)d_in[1];
    const float* difference  = (const float*)d_in[2];
    const float* conv_w      = (const float*)d_in[3];
    const float* conv_b      = (const float*)d_in[4];
    const float* fc1_w       = (const float*)d_in[5];
    const float* fc1_b       = (const float*)d_in[6];
    const float* fc2_w       = (const float*)d_in[7];
    const float* fc2_b       = (const float*)d_in[8];
    const float* key_w       = (const float*)d_in[9];
    const float* key_b       = (const float*)d_in[10];
    const float* sel_w       = (const float*)d_in[11];
    const float* sel_b       = (const float*)d_in[12];
    const float* gumbel      = (const float*)d_in[13];

    // workspace layout (~8.5 MB total)
    unsigned char* ws = (unsigned char*)d_ws;
    _Float16* W1h  = (_Float16*)(ws + 0);                 // 32*1024*2 = 65536 B
    _Float16* W2h  = (_Float16*)(ws + 65536);             // 2048 B
    _Float16* Keyh = (_Float16*)(ws + 67584);             // 2048 B
    _Float16* Selh = (_Float16*)(ws + 69632);             // 2048 B
    float*    Kbuf = (float*)(ws + 73728);                // 8192*32*4  = 1 MiB
    float*    Sbuf = (float*)(ws + 73728 + 1048576);      // 57344*32*4 = 7 MiB

    prep_weights<<<128, 256, 0, stream>>>(fc1_w, fc2_w, key_w, sel_w,
                                          W1h, W2h, Keyh, Selh);

    fused_embed<<<NTILES, 32, 0, stream>>>(self_state, other_state,
                                           conv_w, conv_b,
                                           W1h, fc1_b, W2h, fc2_b,
                                           Keyh, key_b, Selh, sel_b,
                                           Kbuf, Sbuf);

    score_kernel<<<NSELF / 256, 256, 0, stream>>>(Kbuf, Sbuf, difference,
                                                  gumbel, (float*)d_out);
}
